// GaborDownsampleBlock_39788577030789
// MI455X (gfx1250) — matrix operand order repaired and checked by reading, not compiled, not run on hardware
//
#include <hip/hip_runtime.h>
#include <hip/hip_bf16.h>

typedef __attribute__((ext_vector_type(16))) _Float16 v16h;
typedef __attribute__((ext_vector_type(8)))  float    v8f;
typedef int v4i __attribute__((vector_size(4 * sizeof(int))));

#define B_   32
#define CI   64
#define CO   128
#define H_   128
#define W_   128
#define OH_  64
#define OW_  64
#define KSTEPS 32          // K = CI*16 = 1024; 32 K-steps of 32
#define CHUNK  16          // input channels staged per LDS buffer
#define LCOLS  144         // col c <-> iw = c-8 ; row stride 288B (16B aligned)

union BFrag { v16h h; _Float16 e[16]; };

// ---- CDNA5 async global->LDS copies (ASYNCcnt path), with safe fallback ---
#if __has_builtin(__builtin_amdgcn_global_load_async_to_lds_b32)
#define HAVE_ASYNC_LDS 1
#else
#define HAVE_ASYNC_LDS 0
#endif
#if __has_builtin(__builtin_amdgcn_global_load_async_to_lds_b128)
#define HAVE_ASYNC_B128 1
#else
#define HAVE_ASYNC_B128 0
#endif

typedef __attribute__((address_space(1))) int gint_t;
typedef __attribute__((address_space(3))) int lint_t;
typedef __attribute__((address_space(1))) v4i gv4i_t;
typedef __attribute__((address_space(3))) v4i lv4i_t;

__device__ __forceinline__ void async_wait_all() {
#if HAVE_ASYNC_LDS || HAVE_ASYNC_B128
#if __has_builtin(__builtin_amdgcn_s_wait_asynccnt)
  __builtin_amdgcn_s_wait_asynccnt(0);
#else
  asm volatile("s_wait_asynccnt 0" ::: "memory");
#endif
#endif
}

// Copy one 16B-aligned octet of f16 (16 bytes) from global to LDS.
__device__ __forceinline__ void cp_16B(const _Float16* __restrict__ g,
                                       _Float16* __restrict__ l) {
#if HAVE_ASYNC_B128
  __builtin_amdgcn_global_load_async_to_lds_b128(
      (gv4i_t*)(g), (lv4i_t*)(l), /*offset=*/0, /*cpol=*/0);
#elif HAVE_ASYNC_LDS
#pragma unroll
  for (int j = 0; j < 4; ++j)
    __builtin_amdgcn_global_load_async_to_lds_b32(
        (gint_t*)(g + 2 * j), (lint_t*)(l + 2 * j), 0, 0);
#else
#pragma unroll
  for (int j = 0; j < 8; ++j) l[j] = g[j];
#endif
}

// ---------------------------------------------------------------------------
// Kernel 0: one-shot f32 -> f16 conversion of x into workspace.
// Non-temporal f32 reads (streaming) so the 67 MB f16 copy stays L2-resident
// (192 MB L2); the conv kernel's 2x-amplified re-reads then hit L2, not HBM.
// ---------------------------------------------------------------------------
__global__ __launch_bounds__(256)
void x_to_f16_kernel(const float* __restrict__ x, _Float16* __restrict__ xh) {
  const size_t i = ((size_t)blockIdx.x * 256 + threadIdx.x) * 8;
#pragma unroll
  for (int j = 0; j < 8; ++j)
    xh[i + j] = (_Float16)__builtin_nontemporal_load(&x[i + j]);
}

// ---------------------------------------------------------------------------
// Kernel 1: synthesize Gabor filters, pack into WMMA A-fragment layout (f16).
// A-tile is 16(M)x32(K) f16. Layout used here (contiguous k, not the ISA table order):
//   lanes 0-15  : row m = lane,     K in {0..15}
//   lanes 16-31 : row m = lane-16,  K in {16..31}
//   within-lane half index j = klocal & 15
// ---------------------------------------------------------------------------
__global__ void gabor_pack_kernel(const float* __restrict__ freq,
                                  const float* __restrict__ theta,
                                  const float* __restrict__ psi,
                                  const float* __restrict__ sigma,
                                  _Float16* __restrict__ apack,
                                  float* __restrict__ stats) {
  const int t = blockIdx.x * blockDim.x + threadIdx.x;
  if (t < 2 * CO) stats[t] = 0.0f;          // re-zero stats every launch
  if (t >= CO * CI) return;

  const int o = t / CI;
  const int i = t - o * CI;
  const float f  = freq[t];
  const float th = theta[t];
  const float ps = psi[t];
  const float s  = sigma[t];

  float sn, cs;
  __sincosf(th, &sn, &cs);
  const float sp    = s + 0.001f;
  const float inv2s = 0.5f / (sp * sp);
  const float norm  = 1.0f / (2.0f * 3.14f * s * s);
  const float lin[4] = {-1.0f, 0.0f, 1.0f, 2.0f};

  const int mt = o >> 4;
  const int m  = o & 15;

#pragma unroll
  for (int ky = 0; ky < 4; ++ky) {
#pragma unroll
    for (int kx = 0; kx < 4; ++kx) {
      const float X = lin[kx], Y = lin[ky];
      const float rotx =  X * cs + Y * sn;
      const float roty = -X * sn + Y * cs;
      const float g = __expf(-(rotx * rotx + roty * roty) * inv2s) *
                      __cosf(f * rotx + ps) * norm;
      const int k  = i * 16 + ky * 4 + kx;
      const int ks = k >> 5;
      const int kl = k & 31;
      const int hi = kl >> 4;
      const int ln = m + 16 * hi;
      const int j  = kl & 15;
      apack[(((size_t)mt * KSTEPS + ks) * 32 + ln) * 16 + j] = (_Float16)g;
    }
  }
}

// ---------------------------------------------------------------------------
// Stage one 16-channel chunk of the 4-row input band into an LDS buffer.
// Valid data moves in 16B async DMA granules (ASYNCcnt); pad halfs (c=6..7,
// c=136..137) and out-of-range rows are zeroed with DS stores.
// 16ch x 4rows x 16 octets = 1024 transfers -> 4 per thread.
// ---------------------------------------------------------------------------
__device__ __forceinline__ void stage_chunk(const _Float16* __restrict__ xh,
                                            _Float16 (*tile)[4][LCOLS],
                                            int b, int c0, int ihbase, int tid) {
  for (int e = tid; e < CHUNK * 4 * 16; e += 256) {
    const int ch  = e >> 6;          // e / 64
    const int rem = e & 63;
    const int row = rem >> 4;        // 4 rows
    const int oct = rem & 15;        // 16 octets (8 f16 each) per row
    const int iw  = oct * 8;
    const int ih  = ihbase + row;
    _Float16* dst = &tile[ch][row][iw + 8];     // 16B aligned
    if ((unsigned)ih < (unsigned)H_) {
      cp_16B(&xh[(((size_t)b * CI + (c0 + ch)) * H_ + ih) * W_ + iw], dst);
    } else {
#pragma unroll
      for (int j = 0; j < 8; ++j) dst[j] = (_Float16)0.0f;
    }
    if (oct == 0) {                  // c = 6,7  (c=7 <-> iw=-1)
      tile[ch][row][6] = (_Float16)0.0f;
      tile[ch][row][7] = (_Float16)0.0f;
    }
    if (oct == 15) {                 // c = 136,137 (c=136 <-> iw=128)
      tile[ch][row][136] = (_Float16)0.0f;
      tile[ch][row][137] = (_Float16)0.0f;
    }
  }
}

// ---------------------------------------------------------------------------
// Kernel 2: implicit-GEMM conv via v_wmma_f32_16x16x32_f16.
// One block per (b, oh) output row; 8 waves = 2 M-halves x 4 N-subtiles;
// each wave owns 4 M-tiles (64 output channels) x 16 output columns.
// Double-buffered LDS: async DMA for chunk cc+1 overlaps WMMA on chunk cc.
// B-fragment 32(K)x16(N): lanes 0-15 hold K 0-15 (even channel taps),
// lanes 16-31 hold K 16-31 (odd channel taps), column n = lane & 15.
// ---------------------------------------------------------------------------
__global__ __launch_bounds__(256, 1)
void conv_wmma_kernel(const _Float16* __restrict__ xh,
                      const _Float16* __restrict__ apack,
                      float* __restrict__ out) {
  __shared__ _Float16 tile[2][CHUNK][4][LCOLS];   // 2 x 18.4 KB

  const int bid = blockIdx.x;          // 0 .. B_*OH_-1
  const int b   = bid / OH_;
  const int oh  = bid - b * OH_;
  const int tid  = threadIdx.x;
  const int lane = tid & 31;
  const int wave = tid >> 5;
  const int nst  = wave & 3;           // N subtile (16 output columns)
  const int mh   = wave >> 2;          // 0/1 -> M-tiles mh*4 .. mh*4+3
  const int col  = lane & 15;
  const int chSel = lane >> 4;         // 0 -> even channel, 1 -> odd channel
  const int ow0  = nst * 16 + col;
  const int ihbase = 2 * oh - 1;

  v8f acc[4] = {};

  // Prologue: stage chunk 0.
  stage_chunk(xh, tile[0], b, 0, ihbase, tid);
  async_wait_all();
  __syncthreads();

  for (int cc = 0; cc < CI / CHUNK; ++cc) {
    // Kick off async staging of the next chunk into the other buffer.
    if (cc < CI / CHUNK - 1)
      stage_chunk(xh, tile[(cc + 1) & 1], b, (cc + 1) * CHUNK, ihbase, tid);

    _Float16 (*cur)[4][LCOLS] = tile[cc & 1];
#pragma unroll
    for (int ks = 0; ks < CHUNK / 2; ++ks) {
      // B fragment: im2col of channel pair (2ks, 2ks+1) of this chunk.
      BFrag bf;
      const int ch = 2 * ks + chSel;
#pragma unroll
      for (int ky = 0; ky < 4; ++ky) {
#pragma unroll
        for (int kx = 0; kx < 4; ++kx) {
          // iw = 2*ow0 - 1 + kx  ->  c = 2*ow0 + 7 + kx  (c in [7,136])
          bf.e[ky * 4 + kx] = cur[ch][ky][2 * ow0 + 7 + kx];
        }
      }
      const int ksg = cc * (CHUNK / 2) + ks;
#pragma unroll
      for (int i = 0; i < 4; ++i) {
        const int mt = mh * 4 + i;
        const v16h a = *(const v16h*)(apack +
            (((size_t)mt * KSTEPS + ksg) * 32 + lane) * 16);
        acc[i] = __builtin_amdgcn_wmma_f32_16x16x32_f16(
            /*neg_a=*/false, a, /*neg_b=*/false, bf.h,
            /*c_mod=*/(short)0, acc[i], /*reuse_a=*/false, /*reuse_b=*/false);
      }
    }

    async_wait_all();     // next chunk's DMA complete
    __syncthreads();      // visible to all waves; protects buffer reuse
  }

  // C/D layout: VGPR r holds (m = r + 8*(lane>=16), n = lane & 15).
  const int mhi = (lane >> 4) * 8;
#pragma unroll
  for (int i = 0; i < 4; ++i) {
    const int mt = mh * 4 + i;
#pragma unroll
    for (int r = 0; r < 8; ++r) {
      const int o = mt * 16 + r + mhi;
      out[(((size_t)b * CO + o) * OH_ + oh) * OW_ + ow0] = acc[i][r];
    }
  }
}

// ---------------------------------------------------------------------------
// Kernel 3: per-channel sum / sum-of-squares over (B, OH, OW).
// grid = (CO, 8); each block reduces 16384 elements of one channel.
// ---------------------------------------------------------------------------
__global__ __launch_bounds__(256)
void stats_kernel(const float* __restrict__ out, float* __restrict__ stats) {
  __shared__ float ssum[256];
  __shared__ float ssq[256];
  const int ch  = blockIdx.x;
  const int seg = blockIdx.y;
  const int tid = threadIdx.x;

  float s = 0.0f, q = 0.0f;
  const int base = seg * 16384 + tid;
  for (int it = 0; it < 64; ++it) {
    const int j   = base + it * 256;
    const int bb  = j >> 12;
    const int pix = j & 4095;
    const float v = out[(((size_t)bb * CO + ch) << 12) + pix];
    s += v;
    q += v * v;
  }
  ssum[tid] = s;
  ssq[tid]  = q;
  __syncthreads();
  for (int st = 128; st > 0; st >>= 1) {
    if (tid < st) { ssum[tid] += ssum[tid + st]; ssq[tid] += ssq[tid + st]; }
    __syncthreads();
  }
  if (tid == 0) {
    atomicAdd(&stats[ch], ssum[0]);
    atomicAdd(&stats[CO + ch], ssq[0]);
  }
}

// ---------------------------------------------------------------------------
// Kernel 4: in-place batchnorm + affine + leaky ReLU (L2-hot).
// ---------------------------------------------------------------------------
__global__ __launch_bounds__(256)
void bn_lrelu_kernel(float* __restrict__ out, const float* __restrict__ stats,
                     const float* __restrict__ gamma,
                     const float* __restrict__ beta) {
  const float invN = 1.0f / (float)(B_ * OH_ * OW_);
  const size_t i = (size_t)blockIdx.x * 256 + threadIdx.x;
  const int ch = (int)((i >> 12) & (CO - 1));
  const float mean = stats[ch] * invN;
  const float var  = stats[CO + ch] * invN - mean * mean;
  const float sc   = gamma[ch] * rsqrtf(var + 1e-5f);
  const float sh   = beta[ch] - mean * sc;
  const float v    = out[i] * sc + sh;
  out[i] = (v >= 0.0f) ? v : 0.1f * v;
}

extern "C" void kernel_launch(void* const* d_in, const int* in_sizes, int n_in,
                              void* d_out, int out_size, void* d_ws, size_t ws_size,
                              hipStream_t stream) {
  const float* x     = (const float*)d_in[0];
  const float* freq  = (const float*)d_in[1];
  const float* theta = (const float*)d_in[2];
  const float* psi   = (const float*)d_in[3];
  const float* sigma = (const float*)d_in[4];
  const float* gamma = (const float*)d_in[5];
  const float* beta  = (const float*)d_in[6];
  float* out = (float*)d_out;

  _Float16* apack = (_Float16*)d_ws;                         // 256 KB
  float*    stats = (float*)((char*)d_ws + 262144);          // 1 KB
  _Float16* xh    = (_Float16*)((char*)d_ws + 263168);       // 64 MB f16 copy

  const size_t nx = (size_t)B_ * CI * H_ * W_;               // 33.5M elems
  x_to_f16_kernel<<<(unsigned)(nx / (256 * 8)), 256, 0, stream>>>(x, xh);
  gabor_pack_kernel<<<(CO * CI) / 256, 256, 0, stream>>>(freq, theta, psi,
                                                         sigma, apack, stats);
  conv_wmma_kernel<<<B_ * OH_, 256, 0, stream>>>(xh, apack, out);
  stats_kernel<<<dim3(CO, 8), 256, 0, stream>>>(out, stats);
  bn_lrelu_kernel<<<(B_ * CO * OH_ * OW_) / 256, 256, 0, stream>>>(out, stats,
                                                                   gamma, beta);
}
